// BinaryLinear_46926812676154
// MI455X (gfx1250) — compile-verified, matrix operand order checked by reading, not run on hardware
//
#include <hip/hip_runtime.h>
#include <stdint.h>

// ---------------------------------------------------------------------------
// BinaryLinear: out = sign(x) @ sign(W)^T + bias
//   x: [8192,4096] f32, W: [4096,4096] f32, bias: [4096] f32, out f32.
// Exact-math strategy: encode sign as FP8 e4m3 (+1=0x38, -1=0xB8, 0=0x00),
// accumulate with V_WMMA_F32_16X16X128_FP8_FP8 (exact for +/-1 inputs).
// ---------------------------------------------------------------------------

typedef int   v16i __attribute__((ext_vector_type(16)));
typedef int   v2i  __attribute__((ext_vector_type(2)));
typedef float v8f  __attribute__((ext_vector_type(8)));
typedef int   v4i_raw __attribute__((vector_size(16)));   // builtin param type

#define IN_F   4096
#define OUT_F  4096
#define N_TOK  8192

#define BM 128            // block tile M
#define BN 128            // block tile N
#define BK 128            // K bytes staged per step (one fp8 WMMA K)
#define PITCH 144         // LDS row pitch (conflict-free, 16B aligned)
#define A_BYTES (BM * PITCH)
#define STAGE_BYTES (2 * A_BYTES)       // A tile + B tile per stage
#define NKSTEPS (IN_F / BK)             // 32

// ------------------------- CDNA5 async copy path ---------------------------
#if __has_builtin(__builtin_amdgcn_global_load_async_to_lds_b128)
#define HAVE_ASYNC_CP 1
#else
#define HAVE_ASYNC_CP 0
#endif

#if HAVE_ASYNC_CP
static __device__ __forceinline__ void cp16_async(const uint8_t* g, uint8_t* l) {
  // Signature per hipcc diagnostic: generic pointers to gcc-vector int4.
  __builtin_amdgcn_global_load_async_to_lds_b128((v4i_raw*)g, (v4i_raw*)l, 0, 0);
}
#endif

static __device__ __forceinline__ void wait_async_le8() {
#if __has_builtin(__builtin_amdgcn_s_wait_asynccnt)
  __builtin_amdgcn_s_wait_asynccnt(8);
#else
  asm volatile("s_wait_asynccnt 8" ::: "memory");
#endif
}
static __device__ __forceinline__ void wait_async_le0() {
#if __has_builtin(__builtin_amdgcn_s_wait_asynccnt)
  __builtin_amdgcn_s_wait_asynccnt(0);
#else
  asm volatile("s_wait_asynccnt 0" ::: "memory");
#endif
}

// Stage one BK-wide chunk of the A (x) and B (W) tiles into LDS.
// 256 threads move 2*128*128 = 32KB -> 8 x b128 per thread.
static __device__ __forceinline__ void stage(const uint8_t* __restrict__ gA,
                                             const uint8_t* __restrict__ gB,
                                             uint8_t* __restrict__ buf, int t) {
#pragma unroll
  for (int c = 0; c < 4; ++c) {
    const int chunk = t + c * 256;        // 0..1023
    const int row   = chunk >> 3;         // 0..127
    const int off   = (chunk & 7) * 16;   // 0..112
    const uint8_t* ga = gA + (size_t)row * IN_F + off;
    const uint8_t* gb = gB + (size_t)row * IN_F + off;
    uint8_t* la = buf + row * PITCH + off;
    uint8_t* lb = buf + A_BYTES + row * PITCH + off;
#if HAVE_ASYNC_CP
    cp16_async(ga, la);
    cp16_async(gb, lb);
#else
    *(int4*)la = *(const int4*)ga;
    *(int4*)lb = *(const int4*)gb;
#endif
  }
}

// --------------------------- quantize: f32 -> fp8 sign ---------------------
__global__ __launch_bounds__(256)
void BinaryLinear_sign_to_fp8(const float* __restrict__ in,
                              uint8_t* __restrict__ out, int n) {
  const int base = (blockIdx.x * 256 + threadIdx.x) * 16;
  if (base >= n) return;
  const float4* p = (const float4*)(in + base);
  union { uint8_t b[16]; int4 v; } u;
#pragma unroll
  for (int i = 0; i < 4; ++i) {
    const float4 f = p[i];
    const float vals[4] = {f.x, f.y, f.z, f.w};
#pragma unroll
    for (int j = 0; j < 4; ++j) {
      const float v = vals[j];
      u.b[i * 4 + j] = v > 0.f ? 0x38 : (v < 0.f ? 0xB8 : 0x00);  // e4m3 +/-1, 0
    }
  }
  *(int4*)(out + base) = u.v;
}

// Load one 16x128 fp8 fragment (A or B side) from an LDS tile.
// Per the ISA 8-bit layout: lanes 0-15 take byte chunks {0,16,32,48,64,80,96,112},
// lanes 16-31 the same +8; each 8-byte chunk fills one even/odd VGPR pair.
static __device__ __forceinline__ v16i load_frag(const uint8_t* tile, int r16,
                                                 int l16, int hi) {
  const uint8_t* base = tile + (r16 + l16) * PITCH + hi * 8;
  v16i f;
#pragma unroll
  for (int s = 0; s < 8; ++s) {
    const v2i d = *(const v2i*)(base + s * 16);   // ds_load_b64
    f[2 * s]     = d.x;
    f[2 * s + 1] = d.y;
  }
  return f;
}

// ------------------------------- GEMM kernel -------------------------------
__global__ __launch_bounds__(256, 2)
void BinaryLinear_46926812676154_kernel(const uint8_t* __restrict__ xq,
                                        const uint8_t* __restrict__ wq,
                                        const float* __restrict__ bias,
                                        float* __restrict__ out) {
  __shared__ uint8_t smem[2 * STAGE_BYTES];   // double-buffered A+B tiles

  const int t    = threadIdx.x;
  const int bm   = blockIdx.x;                // M tile (64)
  const int bn   = blockIdx.y;                // N tile (32)
  const int wave = t >> 5;
  const int lane = t & 31;
  const int l16  = lane & 15;
  const int hi   = lane >> 4;
  const int wm   = (wave & 3) * 32;           // wave row offset in block tile
  const int wn   = (wave >> 2) * 64;          // wave col offset in block tile

  const uint8_t* gA = xq + (size_t)(bm * BM) * IN_F;
  const uint8_t* gB = wq + (size_t)(bn * BN) * IN_F;

  v8f acc[2][4];
#pragma unroll
  for (int i = 0; i < 2; ++i)
#pragma unroll
    for (int j = 0; j < 4; ++j) acc[i][j] = (v8f)0.f;

#if HAVE_ASYNC_CP
  stage(gA, gB, smem, t);                     // prologue: K-chunk 0 -> buf0
#endif

  for (int kt = 0; kt < NKSTEPS; ++kt) {
    uint8_t* cur = smem + (kt & 1) * STAGE_BYTES;
#if HAVE_ASYNC_CP
    if (kt + 1 < NKSTEPS) {                   // prefetch next chunk
      stage(gA + (kt + 1) * BK, gB + (kt + 1) * BK,
            smem + ((kt + 1) & 1) * STAGE_BYTES, t);
      wait_async_le8();                       // drain the 8 ops of `cur`
    } else {
      wait_async_le0();
    }
    __syncthreads();
#else
    __syncthreads();
    stage(gA + kt * BK, gB + kt * BK, cur, t);
    __syncthreads();
#endif

    const uint8_t* lA = cur;
    const uint8_t* lB = cur + A_BYTES;

    v16i a0 = load_frag(lA, wm, l16, hi);
    v16i a1 = load_frag(lA, wm + 16, l16, hi);
#pragma unroll
    for (int j = 0; j < 4; ++j) {
      const v16i b = load_frag(lB, wn + j * 16, l16, hi);
      acc[0][j] = __builtin_amdgcn_wmma_f32_16x16x128_fp8_fp8(
          a0, b, (short)0, acc[0][j], false, false);
      acc[1][j] = __builtin_amdgcn_wmma_f32_16x16x128_fp8_fp8(
          a1, b, (short)0, acc[1][j], false, false);
    }
    __syncthreads();                          // protect buffer being reloaded
  }

  // Epilogue: C/D 16x16 f32 layout -> VGPR r holds row (r + 8*hi), col = l16.
  const int row0 = bm * BM + wm;
  const int col0 = bn * BN + wn;
#pragma unroll
  for (int j = 0; j < 4; ++j) {
    const int col = col0 + j * 16 + l16;
    const float bv = bias[col];
#pragma unroll
    for (int i = 0; i < 2; ++i) {
#pragma unroll
      for (int r = 0; r < 8; ++r) {
        const int row = row0 + i * 16 + hi * 8 + r;
        out[(size_t)row * OUT_F + col] = acc[i][j][r] + bv;
      }
    }
  }
}

// ------------------------------- launcher ----------------------------------
extern "C" void kernel_launch(void* const* d_in, const int* in_sizes, int n_in,
                              void* d_out, int out_size, void* d_ws, size_t ws_size,
                              hipStream_t stream) {
  const float* x    = (const float*)d_in[0];
  const float* w    = (const float*)d_in[1];
  const float* bias = (const float*)d_in[2];
  float* out        = (float*)d_out;

  uint8_t* xq = (uint8_t*)d_ws;                       // 32 MB
  uint8_t* wq = xq + (size_t)N_TOK * IN_F;            // 16 MB

  const int nx = N_TOK * IN_F;
  const int nw = OUT_F * IN_F;
  BinaryLinear_sign_to_fp8<<<nx / (256 * 16), 256, 0, stream>>>(x, xq, nx);
  BinaryLinear_sign_to_fp8<<<nw / (256 * 16), 256, 0, stream>>>(w, wq, nw);

  dim3 grid(N_TOK / BM, OUT_F / BN);                  // 64 x 32
  BinaryLinear_46926812676154_kernel<<<grid, 256, 0, stream>>>(xq, wq, bias, out);
}
